// GraphTimeModel_80779744903725
// MI455X (gfx1250) — matrix operand order repaired and checked by reading, not compiled, not run on hardware
//
#include <hip/hip_runtime.h>

// ---------------------------------------------------------------------------
// GraphTimeModel forward for MI455X (gfx1250, wave32, WMMA + async-to-LDS).
// ---------------------------------------------------------------------------

typedef __attribute__((ext_vector_type(16))) __bf16 v16bf;
typedef __attribute__((ext_vector_type(8)))  float  v8f;

#define WMMA_BF16(a, b, c) \
  __builtin_amdgcn_wmma_f32_16x16x32_bf16(false, (a), false, (b), (short)0, (c), false, false)

__device__ __forceinline__ float gelu_f(float x) {
  return 0.5f * x * (1.0f + erff(x * 0.70710678118654752f));
}

// CDNA5 async copy: 16B per lane, global -> LDS, tracked with ASYNCcnt.
__device__ __forceinline__ void ld_async_b128(unsigned lds_addr, const void* gaddr) {
  asm volatile("global_load_async_to_lds_b128 %0, %1, off"
               :: "v"(lds_addr), "v"(gaddr) : "memory");
}
__device__ __forceinline__ void wait_async0() {
  asm volatile("s_wait_asynccnt 0x0" ::: "memory");
}

// ---------------- adjacency preprocessing (N=256) ---------------------------
__global__ void k_adj1(const float* __restrict__ L, float* __restrict__ An) {
  int i = blockIdx.x, j = threadIdx.x;
  float a = 0.0f;
  if (j != i) a = gelu_f(L[i * 256 + j] + L[j * 256 + i]);
  __shared__ float red[256];
  red[j] = fabsf(a);
  __syncthreads();
  for (int s = 128; s > 0; s >>= 1) {
    if (j < s) red[j] += red[j + s];
    __syncthreads();
  }
  float denom = red[0] > 1e-12f ? red[0] : 1e-12f;
  An[i * 256 + j] = a / denom;
}

__global__ void k_adj2(const float* __restrict__ An, float* __restrict__ Ain,
                       float* __restrict__ dinv) {
  int i = blockIdx.x, j = threadIdx.x;
  float v = An[j * 256 + i];
  Ain[i * 256 + j] = v;
  __shared__ float red[256];
  red[j] = v;
  __syncthreads();
  for (int s = 128; s > 0; s >>= 1) {
    if (j < s) red[j] += red[j + s];
    __syncthreads();
  }
  if (j == 0) {
    float deg = red[0] + 1.0f;
    deg = deg > 1e-12f ? deg : 1e-12f;
    dinv[i] = rsqrtf(deg);
  }
}

__global__ void k_adj3(const float* __restrict__ Ain, const float* __restrict__ dinv,
                       float* __restrict__ Anorm) {
  int i = blockIdx.x, j = threadIdx.x;
  float v = Ain[i * 256 + j] + (i == j ? 1.0f : 0.0f);
  Anorm[i * 256 + j] = dinv[i] * v * dinv[j];
}

// ---------------- generic WMMA GEMM (fp32 in, bf16 compute, f32 acc) --------
// C[b] = act(A[b](MxK) @ B[b](KxN) + bias). One wave per 16x16 C tile.
__global__ void k_gemm(const float* __restrict__ A, long long sA,
                       const float* __restrict__ B, long long sB,
                       float* __restrict__ C, __bf16* __restrict__ Cbf, long long sC,
                       const float* __restrict__ bias, int act,
                       int M, int N, int K, int lda, int ldb, int ldc) {
  int batch = blockIdx.y;
  const float* Ab = A + (long long)batch * sA;
  const float* Bb = B + (long long)batch * sB;
  int ntN = N >> 4;
  int wave = blockIdx.x * (blockDim.x >> 5) + (threadIdx.x >> 5);
  if (wave >= (M >> 4) * ntN) return;
  int mt = wave / ntN, nt = wave % ntN;
  int lane = threadIdx.x & 31;
  int m0 = mt << 4, n0 = nt << 4;
  int mrow = m0 + (lane & 15);
  int kb = (lane >> 4) << 3;
  v8f acc = {};
  for (int k = 0; k < K; k += 32) {
    v16bf a, b;
    const float* ap = Ab + (long long)mrow * lda + k + kb;
    const float* bp = Bb + (long long)(k + kb) * ldb + n0 + (lane & 15);
    if (k + 32 < K) {                       // global_prefetch_b8 next K panel
      __builtin_prefetch(ap + 32, 0, 1);
      __builtin_prefetch(bp + (long long)32 * ldb, 0, 1);
    }
#pragma unroll
    for (int e = 0; e < 8; ++e) {
      a[e]     = (__bf16)ap[e];
      a[e + 8] = (__bf16)ap[16 + e];
    }
#pragma unroll
    for (int e = 0; e < 8; ++e) {
      b[e]     = (__bf16)bp[(long long)e * ldb];
      b[e + 8] = (__bf16)bp[(long long)(16 + e) * ldb];
    }
    acc = WMMA_BF16(a, b, acc);
  }
  int col = n0 + (lane & 15);
  float bv = bias ? bias[col] : 0.0f;
#pragma unroll
  for (int v = 0; v < 8; ++v) {
    int row = m0 + v + ((lane >> 4) << 3);
    float r = acc[v] + bv;
    if (act) r = gelu_f(r);
    long long idx = (long long)batch * sC + (long long)row * ldc + col;
    if (C)   C[idx]   = r;
    if (Cbf) Cbf[idx] = (__bf16)r;
  }
}

// ---------------- GAT attention scores --------------------------------------
__global__ void k_srcdst(const __bf16* __restrict__ Z, const float* __restrict__ asrc,
                         const float* __restrict__ adst, float* __restrict__ ssrc,
                         float* __restrict__ sdst) {
  int t = blockIdx.x * blockDim.x + threadIdx.x;
  if (t >= 256 * 4 * 256) return;
  int j = t & 255, h = (t >> 8) & 3, bt = t >> 10;
  const __bf16* zp = Z + ((long long)(bt * 256 + j)) * 512 + h * 128;
  float as = 0.0f, ad = 0.0f;
  for (int d = 0; d < 128; ++d) {
    float z = (float)zp[d];
    as += z * asrc[h * 128 + d];
    ad += z * adst[h * 128 + d];
  }
  ssrc[(bt * 4 + h) * 256 + j] = as;
  sdst[(bt * 4 + h) * 256 + j] = ad;
}

// ---------------- fused GAT -------------------------------------------------
// masked softmax in LDS -> async-staged Z panel in LDS -> WMMA alpha@Z.
// block = (bt, itile of 32 rows), 256 threads = 8 waves; 4 heads in registers.
#define ZT_PITCH 136   // 128 bf16 + 8 pad (272B rows, 16B multiple)

__global__ void k_gat(const __bf16* __restrict__ Z, const float* __restrict__ ssrc,
                      const float* __restrict__ sdst, const float* __restrict__ Ain,
                      float* __restrict__ hout) {
  int itile = blockIdx.x & 7;
  int bt = blockIdx.x >> 3;
  int tid = threadIdx.x, wave = tid >> 5, lane = tid & 31;
  __shared__ float ssrc_sh[256];
  __shared__ float sdst_sh[32];
  __shared__ __bf16 alpha_sh[32][256];
  __shared__ __attribute__((aligned(16))) __bf16 ztile[128 * ZT_PITCH];
  unsigned zt_base = (unsigned)(unsigned long long)&ztile[0];
  int srow = tid >> 1;                 // staging: 2 threads per row
  int scb  = (tid & 1) * 128;          // byte offset within 256B row

  v8f acc0 = {}, acc1 = {};
  int n0 = wave << 4;
  int kb = (lane >> 4) << 3;
  int ml = lane & 15;

  for (int h = 0; h < 4; ++h) {
    int bh = bt * 4 + h;
    const __bf16* Zb = Z + ((long long)bt * 256) * 512 + h * 128;
    ssrc_sh[tid] = ssrc[bh * 256 + tid];
    if (tid < 32) sdst_sh[tid] = sdst[bh * 256 + itile * 32 + tid];

    // kick off async DMA of Z half-panel 0 (overlaps with softmax below)
    {
      const char* g = (const char*)(Zb + (long long)srow * 512) + scb;
      unsigned l = zt_base + srow * (ZT_PITCH * 2) + scb;
#pragma unroll
      for (int c = 0; c < 8; ++c) ld_async_b128(l + c * 16, g + c * 16);
    }
    __syncthreads();                   // ssrc_sh/sdst_sh visible

    // masked softmax: wave w owns local rows w*4 .. w*4+3
#pragma unroll
    for (int rr = 0; rr < 4; ++rr) {
      int rl = wave * 4 + rr;
      int i = itile * 32 + rl;
      float sd = sdst_sh[rl];
      float vals[8];
      int   msk[8];
      float vmax = -3.4e38f;
#pragma unroll
      for (int jj = 0; jj < 8; ++jj) {
        int j = jj * 32 + lane;
        float v = sd + ssrc_sh[j];
        v = v > 0.0f ? v : 0.2f * v;   // leaky_relu 0.2
        int m = (Ain[i * 256 + j] != 0.0f) || (i == j);
        vals[jj] = v; msk[jj] = m;
        if (m && v > vmax) vmax = v;
      }
#pragma unroll
      for (int o = 16; o > 0; o >>= 1) {
        float tmx = __shfl_xor(vmax, o, 32);
        vmax = tmx > vmax ? tmx : vmax;
      }
      float es[8], sum = 0.0f;
#pragma unroll
      for (int jj = 0; jj < 8; ++jj) {
        float e = msk[jj] ? __expf(vals[jj] - vmax) : 0.0f;
        es[jj] = e; sum += e;
      }
#pragma unroll
      for (int o = 16; o > 0; o >>= 1) sum += __shfl_xor(sum, o, 32);
      float inv = 1.0f / sum;          // diagonal always unmasked
#pragma unroll
      for (int jj = 0; jj < 8; ++jj)
        alpha_sh[rl][jj * 32 + lane] = (__bf16)(es[jj] * inv);
    }

    // two K halves: WMMA consumes LDS panel, restage second half in between
#pragma unroll
    for (int half = 0; half < 2; ++half) {
      wait_async0();
      __syncthreads();                 // panel staged + alpha written
#pragma unroll
      for (int k = 0; k < 128; k += 32) {
        v16bf a0, a1, b;
        int kg = half * 128 + k;
#pragma unroll
        for (int e = 0; e < 8; ++e) {
          a0[e]     = alpha_sh[ml][kg + kb + e];
          a0[e + 8] = alpha_sh[ml][kg + 16 + kb + e];
          a1[e]     = alpha_sh[16 + ml][kg + kb + e];
          a1[e + 8] = alpha_sh[16 + ml][kg + 16 + kb + e];
        }
        const __bf16* zr = ztile + (k + kb) * ZT_PITCH + n0 + ml;
#pragma unroll
        for (int e = 0; e < 8; ++e) {
          b[e]     = zr[e * ZT_PITCH];
          b[e + 8] = zr[(16 + e) * ZT_PITCH];
        }
        acc0 = WMMA_BF16(a0, b, acc0);
        acc1 = WMMA_BF16(a1, b, acc1);
      }
      __syncthreads();                 // all waves done reading ztile
      if (half == 0) {                 // stage rows 128..255
        const char* g = (const char*)(Zb + (long long)(128 + srow) * 512) + scb;
        unsigned l = zt_base + srow * (ZT_PITCH * 2) + scb;
#pragma unroll
        for (int c = 0; c < 8; ++c) ld_async_b128(l + c * 16, g + c * 16);
      }
    }
    // trailing __syncthreads of half loop protects alpha_sh/ssrc_sh reuse
  }

  int col = n0 + ml;
#pragma unroll
  for (int v = 0; v < 8; ++v) {
    int r0 = itile * 32 + v + ((lane >> 4) << 3);
    hout[((long long)bt * 256 + r0) * 128 + col]      = acc0[v] * 0.25f;
    hout[((long long)bt * 256 + r0 + 16) * 128 + col] = acc1[v] * 0.25f;
  }
}

// h = gelu(h + b_gat) over (65536,128)
__global__ void k_bias_gelu(float* __restrict__ h, const float* __restrict__ b,
                            long long n) {
  long long i = (long long)blockIdx.x * 256 + threadIdx.x;
  if (i < n) h[i] = gelu_f(h[i] + b[i & 127]);
}

// h_seq[bt,d] = mean_n h[bt,n,d]
__global__ void k_meanpool(const float* __restrict__ h, float* __restrict__ hseq) {
  int bt = blockIdx.x, d = threadIdx.x;
  const float* p = h + (long long)bt * 256 * 128 + d;
  float s = 0.0f;
  for (int n = 0; n < 256; ++n) s += p[n * 128];
  hseq[bt * 128 + d] = s * (1.0f / 256.0f);
}

// ---------------- GRU scan: recurrent part only (GI precomputed by WMMA) ----
// one block per batch, 384 threads (one per gate unit).
__global__ void k_gru_scan(const float* __restrict__ GI, const float* __restrict__ Wh,
                           const float* __restrict__ bh, float* __restrict__ gruo) {
  int b = blockIdx.x;          // 16
  int g = threadIdx.x;         // 384
  __shared__ float hprev[128];
  __shared__ float gh_sh[384];
  if (g < 128) hprev[g] = 0.0f;
  __syncthreads();
  for (int t = 0; t < 16; ++t) {
    float s = bh[g];
    for (int d = 0; d < 128; ++d) s += hprev[d] * Wh[d * 384 + g];
    gh_sh[g] = s;
    __syncthreads();
    if (g < 128) {
      const float* gi = GI + (b * 16 + t) * 384;
      float r = 1.0f / (1.0f + __expf(-(gi[g] + gh_sh[g])));
      float z = 1.0f / (1.0f + __expf(-(gi[128 + g] + gh_sh[128 + g])));
      float n = tanhf(gi[256 + g] + r * gh_sh[256 + g]);
      float hn = (1.0f - z) * n + z * hprev[g];
      gruo[(b * 16 + t) * 128 + g] = hn;
      hprev[g] = hn;
    }
    __syncthreads();
  }
}

// ---------------- attention + fusion + classifier, single block -------------
__global__ void k_head(const float* __restrict__ gruo, const float* __restrict__ watt,
                       const float* __restrict__ batt, const float* __restrict__ ln1g,
                       const float* __restrict__ ln1b, const float* __restrict__ Wf,
                       const float* __restrict__ bf_, const float* __restrict__ Wc1,
                       const float* __restrict__ bc1, const float* __restrict__ ln2g,
                       const float* __restrict__ ln2b, const float* __restrict__ Wc2,
                       const float* __restrict__ bc2, float* __restrict__ out) {
  __shared__ float wlog[256];
  __shared__ float att[16 * 128];
  __shared__ float fin[16 * 128];
  __shared__ float c1[16 * 256];
  __shared__ float mu[16], rs[16];
  int tid = threadIdx.x;
  {
    int b = tid >> 4, t = tid & 15;
    const float* g = gruo + (b * 16 + t) * 128;
    float s = batt[0];
    for (int d = 0; d < 128; ++d) s += g[d] * watt[d];
    wlog[tid] = s;
  }
  __syncthreads();
  if (tid < 16) {
    float mx = -3.4e38f;
    for (int t = 0; t < 16; ++t) mx = fmaxf(mx, wlog[tid * 16 + t]);
    float s = 0.0f;
    for (int t = 0; t < 16; ++t) { float e = __expf(wlog[tid * 16 + t] - mx); wlog[tid * 16 + t] = e; s += e; }
    float inv = 1.0f / s;
    for (int t = 0; t < 16; ++t) wlog[tid * 16 + t] *= inv;
  }
  __syncthreads();
  for (int p = tid; p < 16 * 128; p += 256) {
    int b = p >> 7, d = p & 127;
    float s = 0.0f, m = 0.0f;
    for (int t = 0; t < 16; ++t) {
      float g = gruo[(b * 16 + t) * 128 + d];
      s += wlog[b * 16 + t] * g;
      m += g;
    }
    att[p] = s + m * (1.0f / 16.0f);
  }
  __syncthreads();
  if (tid < 16) {
    float m = 0.0f;
    for (int d = 0; d < 128; ++d) m += att[tid * 128 + d];
    m *= (1.0f / 128.0f);
    float v = 0.0f;
    for (int d = 0; d < 128; ++d) { float x = att[tid * 128 + d] - m; v += x * x; }
    mu[tid] = m; rs[tid] = rsqrtf(v * (1.0f / 128.0f) + 1e-5f);
  }
  __syncthreads();
  for (int p = tid; p < 16 * 128; p += 256) {
    int b = p >> 7, d = p & 127;
    att[p] = (att[p] - mu[b]) * rs[b] * ln1g[d] + ln1b[d];
  }
  __syncthreads();
  for (int p = tid; p < 16 * 128; p += 256) {
    int b = p >> 7, k = p & 127;
    const float* gl = gruo + (b * 16 + 15) * 128;
    float s = bf_[k];
    for (int d = 0; d < 128; ++d) s += att[b * 128 + d] * Wf[d * 128 + k];
    for (int d = 0; d < 128; ++d) s += gl[d] * Wf[(128 + d) * 128 + k];
    fin[p] = s;
  }
  __syncthreads();
  for (int p = tid; p < 16 * 256; p += 256) {
    int b = p >> 8, k = p & 255;
    float s = bc1[k];
    for (int d = 0; d < 128; ++d) s += fin[b * 128 + d] * Wc1[d * 256 + k];
    c1[p] = s;
  }
  __syncthreads();
  if (tid < 16) {
    float m = 0.0f;
    for (int d = 0; d < 256; ++d) m += c1[tid * 256 + d];
    m *= (1.0f / 256.0f);
    float v = 0.0f;
    for (int d = 0; d < 256; ++d) { float x = c1[tid * 256 + d] - m; v += x * x; }
    mu[tid] = m; rs[tid] = rsqrtf(v * (1.0f / 256.0f) + 1e-5f);
  }
  __syncthreads();
  for (int p = tid; p < 16 * 256; p += 256) {
    int b = p >> 8, d = p & 255;
    c1[p] = gelu_f((c1[p] - mu[b]) * rs[b] * ln2g[d] + ln2b[d]);
  }
  __syncthreads();
  for (int p = tid; p < 160; p += 256) {
    int b = p / 10, c = p % 10;
    float s = bc2[c];
    for (int d = 0; d < 256; ++d) s += c1[b * 256 + d] * Wc2[d * 10 + c];
    out[b * 10 + c] = s;
  }
}

// ---------------------------------------------------------------------------
extern "C" void kernel_launch(void* const* d_in, const int* in_sizes, int n_in,
                              void* d_out, int out_size, void* d_ws, size_t ws_size,
                              hipStream_t stream) {
  (void)in_sizes; (void)n_in; (void)out_size; (void)ws_size;
  const float* x    = (const float*)d_in[0];
  const float* Ladj = (const float*)d_in[2];
  const float* Wgat = (const float*)d_in[3];
  const float* asrc = (const float*)d_in[4];
  const float* adst = (const float*)d_in[5];
  const float* bgat = (const float*)d_in[6];
  const float* Wgcn = (const float*)d_in[7];
  const float* bgcn = (const float*)d_in[8];
  const float* Wi   = (const float*)d_in[9];
  const float* Wh   = (const float*)d_in[10];
  const float* bi   = (const float*)d_in[11];
  const float* bh   = (const float*)d_in[12];
  const float* watt = (const float*)d_in[13];
  const float* batt = (const float*)d_in[14];
  const float* ln1g = (const float*)d_in[15];
  const float* ln1b = (const float*)d_in[16];
  const float* Wf   = (const float*)d_in[17];
  const float* bf_  = (const float*)d_in[18];
  const float* Wc1  = (const float*)d_in[19];
  const float* bc1  = (const float*)d_in[20];
  const float* ln2g = (const float*)d_in[21];
  const float* ln2b = (const float*)d_in[22];
  const float* Wc2  = (const float*)d_in[23];
  const float* bc2  = (const float*)d_in[24];

  char* ws = (char*)d_ws;
  size_t off = 0;
  auto carve = [&](size_t bytes) {
    void* p = ws + off;
    off = (off + bytes + 255) & ~(size_t)255;
    return p;
  };
  float*  An    = (float*)carve(256 * 256 * 4);
  float*  Ain   = (float*)carve(256 * 256 * 4);
  float*  dinv  = (float*)carve(256 * 4);
  float*  Anorm = (float*)carve(256 * 256 * 4);
  __bf16* Zbf   = (__bf16*)carve((size_t)65536 * 512 * 2);
  float*  ssrc  = (float*)carve((size_t)256 * 4 * 256 * 4);
  float*  sdst  = (float*)carve((size_t)256 * 4 * 256 * 4);
  float*  bufA  = (float*)carve((size_t)65536 * 128 * 4);
  float*  bufB  = (float*)carve((size_t)65536 * 128 * 4);
  float*  bufC  = (float*)carve((size_t)65536 * 128 * 4);
  float*  hseq  = (float*)carve(16 * 16 * 128 * 4);
  float*  GI    = (float*)carve(256 * 384 * 4);
  float*  gruo  = (float*)carve(16 * 16 * 128 * 4);

  // adjacency pipeline
  k_adj1<<<256, 256, 0, stream>>>(Ladj, An);
  k_adj2<<<256, 256, 0, stream>>>(An, Ain, dinv);
  k_adj3<<<256, 256, 0, stream>>>(Ain, dinv, Anorm);

  // Z = x @ W_gat -> bf16  (M=65536, N=512, K=64)
  k_gemm<<<dim3(16384, 1), 256, 0, stream>>>(x, 0, Wgat, 0, nullptr, Zbf, 0,
                                             nullptr, 0, 65536, 512, 64, 64, 512, 512);
  k_srcdst<<<1024, 256, 0, stream>>>(Zbf, asrc, adst, ssrc, sdst);
  // fused GAT: async-staged Z panels + LDS softmax + WMMA, head-mean in regs
  k_gat<<<2048, 256, 0, stream>>>(Zbf, ssrc, sdst, Ain, bufA);
  k_bias_gelu<<<32768, 256, 0, stream>>>(bufA, bgat, (long long)65536 * 128);

  // GCN layer 1
  k_gemm<<<dim3(4096, 1), 256, 0, stream>>>(bufA, 0, Wgcn, 0, bufB, nullptr, 0,
                                            nullptr, 0, 65536, 128, 128, 128, 128, 128);
  k_gemm<<<dim3(16, 256), 256, 0, stream>>>(Anorm, 0, bufB, (long long)256 * 128,
                                            bufC, nullptr, (long long)256 * 128,
                                            bgcn, 1, 256, 128, 256, 256, 128, 128);
  // GCN layer 2
  k_gemm<<<dim3(4096, 1), 256, 0, stream>>>(bufC, 0, Wgcn, 0, bufB, nullptr, 0,
                                            nullptr, 0, 65536, 128, 128, 128, 128, 128);
  k_gemm<<<dim3(16, 256), 256, 0, stream>>>(Anorm, 0, bufB, (long long)256 * 128,
                                            bufA, nullptr, (long long)256 * 128,
                                            bgcn, 1, 256, 128, 256, 256, 128, 128);

  // temporal pipeline
  k_meanpool<<<256, 128, 0, stream>>>(bufA, hseq);
  // GI = h_seq @ Wi + bi via WMMA (M=256, N=384, K=128): 384 tiles / 8 waves
  k_gemm<<<dim3(48, 1), 256, 0, stream>>>(hseq, 0, Wi, 0, GI, nullptr, 0,
                                          bi, 0, 256, 384, 128, 128, 384, 384);
  k_gru_scan<<<16, 384, 0, stream>>>(GI, Wh, bh, gruo);
  k_head<<<1, 256, 0, stream>>>(gruo, watt, batt, ln1g, ln1b, Wf, bf_,
                                Wc1, bc1, ln2g, ln2b, Wc2, bc2, (float*)d_out);
}
